// StackedCGLSTM_37830071943201
// MI455X (gfx1250) — matrix operand order repaired and checked by reading, not compiled, not run on hardware
//
#include <hip/hip_runtime.h>
#include <stdint.h>
#include <stddef.h>

// ---------------------------------------------------------------------------
// StackedCGLSTM for MI455X (gfx1250, wave32, WMMA bf16).
//   L=3, B=128, T=256, D=H=F=512.
// Strategy:
//   1) Convert weights/activations to bf16 once per call.
//   2) Two big input GEMMs (M = B*T = 32768) done with v_wmma_f32_16x16x32_bf16:
//        xz[t,b,0:2048]  = |x| @ Wb[0][:,512:1024]^T        (layer-0 x part)
//        gp[t,b,0:6144]  = g_shift @ Wg^T  (f-gate block pre-scaled by -0.5)
//   3) 768 sequential (t,l) fused step kernels: WMMA GEMM over [h_prev, inp]
//      (K=512 for l==0, 1024 otherwise) + gate math + cell update, writing
//      bf16 h into a ping-pong buffer and f32 output at (t=T-1, l=L-1).
// Workspace: ~1.16 GB laid out below.
// ---------------------------------------------------------------------------

#define LNUM 3
#define BNUM 128
#define TNUM 256
#define DNUM 512
#define HNUM 512
#define FNUM 512
#define G4H  2048   // 4*H
#define NGP  6144   // L*4*H
#define MROWS (BNUM * TNUM)   // 32768

typedef __attribute__((ext_vector_type(16))) __bf16 v16bf;
typedef __attribute__((ext_vector_type(8)))  float  v8f;

union FragBF { v16bf v; uint32_t w[8]; };

__device__ __forceinline__ uint16_t f2bf(float f) {
    uint32_t u = __float_as_uint(f);
    u += 0x7FFFu + ((u >> 16) & 1u);   // round-to-nearest-even
    return (uint16_t)(u >> 16);
}
__device__ __forceinline__ float sigm(float x) { return 1.0f / (1.0f + __expf(-x)); }

// -------------------------- conversion kernels ------------------------------

__global__ void k_f32_to_bf16(const float* __restrict__ in,
                              uint16_t* __restrict__ out, int n) {
    int i = blockIdx.x * blockDim.x + threadIdx.x;
    if (i < n) out[i] = f2bf(in[i]);
}

// xabs[(t*B + b)*D + d] = bf16(|hrrp[b][t][d]|)
__global__ void k_abs_x(const float* __restrict__ hrrp,
                        uint16_t* __restrict__ xabs) {
    int i  = blockIdx.x * blockDim.x + threadIdx.x;   // over T*B*D
    int d  = i & (DNUM - 1);
    int bt = i >> 9;
    int b  = bt & (BNUM - 1);
    int t  = bt >> 7;
    xabs[i] = f2bf(fabsf(hrrp[((size_t)b * TNUM + t) * DNUM + d]));
}

// gsh[(t*B + b)*F + f] = (t==0) ? 0 : bf16(gamma[b][t-1][f])
__global__ void k_shift_g(const float* __restrict__ gamma,
                          uint16_t* __restrict__ gsh) {
    int i  = blockIdx.x * blockDim.x + threadIdx.x;   // over T*B*F
    int f  = i & (FNUM - 1);
    int bt = i >> 9;
    int b  = bt & (BNUM - 1);
    int t  = bt >> 7;
    gsh[i] = (t == 0) ? (uint16_t)0
                      : f2bf(gamma[((size_t)b * TNUM + (t - 1)) * FNUM + f]);
}

__global__ void k_init_state(uint16_t* __restrict__ hA,
                             uint16_t* __restrict__ hB,
                             float* __restrict__ c, int n) {
    int i = blockIdx.x * blockDim.x + threadIdx.x;
    if (i < n) { hA[i] = 0; hB[i] = 0; c[i] = 0.0f; }
}

// ------------------------ big input-GEMM (bf16 WMMA) ------------------------
// C[m,n] = sum_k A[m,k] * W[n,k].  A:[M,K] bf16 lda, W:[N,K] bf16 ldw.
// grid.x = M/32, grid.y = N/256, block = 256 (8 waves; each wave: 16m x 64n).
__global__ void k_gemm_bf16_nt(const uint16_t* __restrict__ A, int lda,
                               const uint16_t* __restrict__ W, int ldw,
                               float* __restrict__ C, int ldc,
                               int K, int fgate_scale) {
    const int lane  = threadIdx.x & 31;
    const int wave  = threadIdx.x >> 5;
    const int wm    = wave >> 2, wn = wave & 3;
    const int m0    = blockIdx.x * 32 + wm * 16;
    const int nb    = blockIdx.y * 256 + wn * 64;
    const int rowA  = m0 + (lane & 15);
    const int kbA   = (lane >> 4) * 8;    // A-frag K base (ISA 16-bit A layout)
    const int kbB   = (lane >> 4) * 8;    // B-frag K base (same K order as A)

    v8f acc0 = {}, acc1 = {}, acc2 = {}, acc3 = {};
    for (int kk = 0; kk < K; kk += 32) {
        FragBF a;
        const uint32_t* pa =
            (const uint32_t*)(A + (size_t)rowA * lda + kk + kbA);
#pragma unroll
        for (int j = 0; j < 4; ++j) { a.w[j] = pa[j]; a.w[j + 4] = pa[j + 8]; }

#pragma unroll
        for (int t4 = 0; t4 < 4; ++t4) {
            const int col = nb + t4 * 16 + (lane & 15);
            FragBF b;
            const uint32_t* pb =
                (const uint32_t*)(W + (size_t)col * ldw + kk + kbB);
#pragma unroll
            for (int j = 0; j < 8; ++j) b.w[j] = pb[(j & 3) + (j >> 2) * 8];
            v8f& acc = (t4 == 0) ? acc0 : (t4 == 1) ? acc1 : (t4 == 2) ? acc2 : acc3;
            acc = __builtin_amdgcn_wmma_f32_16x16x32_bf16(
                false, a.v, false, b.v, (short)0, acc, false, false);
        }
    }

    const int mrow = m0 + (lane >> 4) * 8;
#pragma unroll
    for (int r = 0; r < 8; ++r) {
        const int m = mrow + r;
#pragma unroll
        for (int t4 = 0; t4 < 4; ++t4) {
            const int n = nb + t4 * 16 + (lane & 15);
            float v = ((t4 == 0) ? acc0 : (t4 == 1) ? acc1 : (t4 == 2) ? acc2 : acc3)[r];
            if (fgate_scale && (((n & (G4H - 1)) >> 9) == 1)) v *= -0.5f; // -LAM on f-gate
            C[(size_t)m * ldc + n] = v;
        }
    }
}

// ---------------------- fused recurrent step kernel -------------------------
// One launch per (t, l). block = 128 (4 waves), grid = (8, 8).
// Wave computes a 16(b) x 16(h) tile for all 4 gates (4 accumulators),
// then does the full LSTM cell update in the epilogue.
__global__ void k_lstm_step(const uint16_t* __restrict__ hprev, // [L][B][H] bf16
                            uint16_t* __restrict__ hcur,        // [L][B][H] bf16
                            float* __restrict__ cstate,         // [L][B][H] f32
                            const uint16_t* __restrict__ wb_l,  // [4H][H+D] bf16
                            const float* __restrict__ bb_l,     // [4H]
                            const float* __restrict__ gp_tl,    // + b*NGP, gate-major
                            const float* __restrict__ xz_t,     // + b*4H (l==0 only)
                            float* __restrict__ out,            // non-null at final
                            int l, int kmax) {
    const int lane = threadIdx.x & 31;
    const int wave = threadIdx.x >> 5;
    const int h0   = (blockIdx.y * 4 + wave) * 16;
    const int b0   = blockIdx.x * 16;
    const int rowB = b0 + (lane & 15);
    const int kbA  = (lane >> 4) * 8;
    const int kbB  = (lane >> 4) * 8;

    v8f acc0 = {}, acc1 = {}, acc2 = {}, acc3 = {};
    for (int kk = 0; kk < kmax; kk += 32) {
        // A operand: cat(h_prev[l], inp) where inp = hcur[l-1] (current t).
        const uint16_t* asrc =
            (kk < HNUM)
                ? hprev + ((size_t)l * BNUM + rowB) * HNUM + kk
                : hcur + ((size_t)(l - 1) * BNUM + rowB) * HNUM + (kk - HNUM);
        FragBF a;
        const uint32_t* pa = (const uint32_t*)(asrc + kbA);
#pragma unroll
        for (int j = 0; j < 4; ++j) { a.w[j] = pa[j]; a.w[j + 4] = pa[j + 8]; }

#pragma unroll
        for (int g = 0; g < 4; ++g) {
            const int jcol = g * HNUM + h0 + (lane & 15);
            FragBF b;
            const uint32_t* pb =
                (const uint32_t*)(wb_l + (size_t)jcol * (HNUM + DNUM) + kk + kbB);
#pragma unroll
            for (int j = 0; j < 8; ++j) b.w[j] = pb[(j & 3) + (j >> 2) * 8];
            v8f& acc = (g == 0) ? acc0 : (g == 1) ? acc1 : (g == 2) ? acc2 : acc3;
            acc = __builtin_amdgcn_wmma_f32_16x16x32_bf16(
                false, a.v, false, b.v, (short)0, acc, false, false);
        }
    }

    const int brow = b0 + (lane >> 4) * 8;
    const int h    = h0 + (lane & 15);
#pragma unroll
    for (int r = 0; r < 8; ++r) {
        const int b = brow + r;
        const float* gpb = gp_tl + (size_t)b * NGP;
        float zi = acc0[r] + bb_l[h]            + gpb[h];
        float zf = acc1[r] + bb_l[HNUM + h]     + gpb[HNUM + h];     // -0.5*gp baked in
        float zc = acc2[r] + bb_l[2 * HNUM + h] + gpb[2 * HNUM + h];
        float zo = acc3[r] + bb_l[3 * HNUM + h] + gpb[3 * HNUM + h];
        if (xz_t) {
            const float* xb = xz_t + (size_t)b * G4H;
            zi += xb[h]; zf += xb[HNUM + h];
            zc += xb[2 * HNUM + h]; zo += xb[3 * HNUM + h];
        }
        const float it = sigm(zi);
        const float ft = sigm(zf);
        const float ch = tanhf(zc);
        const float ot = sigm(zo);
        const size_t idx = ((size_t)l * BNUM + b) * HNUM + h;
        const float cn = ft * cstate[idx] + it * ch;
        cstate[idx] = cn;
        const float hn = ot * tanhf(cn);
        hcur[idx] = f2bf(hn);
        if (out) out[(size_t)b * HNUM + h] = hn;
    }
}

// ------------------------------- host side ----------------------------------

extern "C" void kernel_launch(void* const* d_in, const int* in_sizes, int n_in,
                              void* d_out, int out_size, void* d_ws, size_t ws_size,
                              hipStream_t stream) {
    (void)in_sizes; (void)n_in; (void)out_size; (void)ws_size;
    const float* hrrp  = (const float*)d_in[0];   // [B,T,D]
    const float* gamma = (const float*)d_in[1];   // [B,T,F]
    const float* Wb    = (const float*)d_in[2];   // [L,4H,H+D]
    const float* bb    = (const float*)d_in[3];   // [L,4H]
    const float* Wg    = (const float*)d_in[4];   // [L,4,H,F]
    float* out = (float*)d_out;                   // [B,H]

    // Workspace layout (256B aligned).
    char* base = (char*)d_ws;
    size_t off = 0;
    auto take = [&](size_t bytes) -> char* {
        char* p = base + off;
        off = (off + bytes + 255) & ~(size_t)255;
        return p;
    };
    uint16_t* wb_bf = (uint16_t*)take((size_t)LNUM * G4H * (HNUM + DNUM) * 2); // 12.6 MB
    uint16_t* wg_bf = (uint16_t*)take((size_t)NGP * FNUM * 2);                 //  6.3 MB
    uint16_t* xabs  = (uint16_t*)take((size_t)MROWS * DNUM * 2);               // 33.6 MB
    uint16_t* gsh   = (uint16_t*)take((size_t)MROWS * FNUM * 2);               // 33.6 MB
    float*    xz    = (float*)take((size_t)MROWS * G4H * 4);                   // 268 MB
    float*    gp    = (float*)take((size_t)MROWS * NGP * 4);                   // 805 MB
    uint16_t* hA    = (uint16_t*)take((size_t)LNUM * BNUM * HNUM * 2);
    uint16_t* hB    = (uint16_t*)take((size_t)LNUM * BNUM * HNUM * 2);
    float*    cst   = (float*)take((size_t)LNUM * BNUM * HNUM * 4);

    // 1) Conversions / init.
    {
        int n = LNUM * G4H * (HNUM + DNUM);
        k_f32_to_bf16<<<n / 256, 256, 0, stream>>>(Wb, wb_bf, n);
    }
    {
        int n = NGP * FNUM;
        k_f32_to_bf16<<<n / 256, 256, 0, stream>>>(Wg, wg_bf, n);
    }
    k_abs_x<<<(MROWS * DNUM) / 256, 256, 0, stream>>>(hrrp, xabs);
    k_shift_g<<<(MROWS * FNUM) / 256, 256, 0, stream>>>(gamma, gsh);
    {
        int n = LNUM * BNUM * HNUM;
        k_init_state<<<n / 256, 256, 0, stream>>>(hA, hB, cst, n);
    }

    // 2) Big input GEMMs.
    // xz = |x| @ Wb[0][:,512:1024]^T  (W rows offset by 512 columns, ldw=1024)
    k_gemm_bf16_nt<<<dim3(MROWS / 32, G4H / 256), 256, 0, stream>>>(
        xabs, DNUM, wb_bf + DNUM, HNUM + DNUM, xz, G4H, FNUM, /*fgate=*/0);
    // gp = g_shift @ Wg^T  (N = 6144; f-gate block scaled by -0.5)
    k_gemm_bf16_nt<<<dim3(MROWS / 32, NGP / 256), 256, 0, stream>>>(
        gsh, FNUM, wg_bf, FNUM, gp, NGP, FNUM, /*fgate=*/1);

    // 3) Sequential recurrence: 256 steps x 3 layers, ping-pong h buffers.
    for (int t = 0; t < TNUM; ++t) {
        uint16_t* hp = (t & 1) ? hB : hA;
        uint16_t* hc = (t & 1) ? hA : hB;
        const float* gp_t = gp + (size_t)t * BNUM * NGP;
        const float* xz_t = xz + (size_t)t * BNUM * G4H;
        for (int l = 0; l < LNUM; ++l) {
            const uint16_t* wb_l = wb_bf + (size_t)l * G4H * (HNUM + DNUM);
            const float*    bb_l = bb + (size_t)l * G4H;
            const float*    gp_tl = gp_t + (size_t)l * G4H;
            float* ow = (t == TNUM - 1 && l == LNUM - 1) ? out : nullptr;
            k_lstm_step<<<dim3(BNUM / 16, (HNUM / 16) / 4), 128, 0, stream>>>(
                hp, hc, cst, wb_l, bb_l, gp_tl,
                (l == 0) ? xz_t : nullptr, ow, l,
                (l == 0) ? HNUM : (HNUM + DNUM));
        }
    }
}
